// MultiHeadAttention_4217657885188
// MI455X (gfx1250) — matrix operand order repaired and checked by reading, not compiled, not run on hardware
//
#include <hip/hip_runtime.h>
#include <hip/hip_bf16.h>

// Problem constants (match reference)
#define BQ    2
#define NSEQ  2048
#define DM    1024
#define NH    16
#define DK    64
#define MTOK  (BQ * NSEQ)   // 4096 tokens

typedef _Float16 v8h  __attribute__((ext_vector_type(8)));
typedef _Float16 v16h __attribute__((ext_vector_type(16)));
typedef float    v8f  __attribute__((ext_vector_type(8)));

union V16H { v16h v; v8h h[2]; };

// A-fragment (16x32 f16, M x K): lane = row (lane&15); lane half selects ko=0/8;
// halves h[0..7] = K[ko..ko+7], h[8..15] = K[16+ko..23+ko].  (ISA 7.12.2)
__device__ __forceinline__ v16h load_a_frag(const _Float16* __restrict__ rowk, int lane) {
  const int ko = (lane >> 4) * 8;
  V16H f;
  f.h[0] = *(const v8h*)(rowk + ko);
  f.h[1] = *(const v8h*)(rowk + 16 + ko);
  return f.v;
}

// B-fragment (32x16 f16, K x N): lane = column (lane&15); lane half selects ko=0/8;
// halves h[0..7] = K[ko..ko+7], h[8..15] = K[16+ko..23+ko] (same as A).
__device__ __forceinline__ v16h load_b_frag(const _Float16* __restrict__ colk, int lane) {
  const int off = (lane >> 4) * 8;
  V16H f;
  f.h[0] = *(const v8h*)(colk + off);
  f.h[1] = *(const v8h*)(colk + 16 + off);
  return f.v;
}

#define WMMA_F16(a, b, c) \
  __builtin_amdgcn_wmma_f32_16x16x32_f16(false, (a), false, (b), (short)0, (c), false, false)

// ---------------------------------------------------------------- f32 -> f16
__global__ __launch_bounds__(256) void cvt_f32_to_f16(const float* __restrict__ src,
                                                      _Float16* __restrict__ dst, int n) {
  int i = blockIdx.x * blockDim.x + threadIdx.x;
  if (i < n) dst[i] = (_Float16)src[i];
}

// -------------------------------------------- register-blocked WMMA GEMM
// C = A @ W^T + bias. Each wave computes a 64x32 tile (4x2 WMMA tiles):
// per K=32 chunk: 4 A-frags + 2 B-frags -> 8 v_wmma (8 accumulators, 64 VGPRs).
// A: [4096 x 1024] f16 row-major; W: [1024 x 1024] f16 row-major.
// mode 0: out16[((b*16+h)*2048 + tok)*64 + dk]   (Q, K head layout)
// mode 1: out16[((b*16+h)*64 + dk)*2048 + tok]   (V transposed head layout)
// mode 2: outf[m*1024 + n]                        (final fp32 output)
__global__ __launch_bounds__(128) void gemm64x32_f16(const _Float16* __restrict__ A,
                                                     const _Float16* __restrict__ W,
                                                     const float* __restrict__ bias,
                                                     _Float16* __restrict__ out16,
                                                     float* __restrict__ outf,
                                                     int mode) {
  const int lane = threadIdx.x & 31;
  const int wave = threadIdx.x >> 5;
  const int ln   = lane & 15;
  const int hi   = (lane >> 4) * 8;

  const int NT   = DM / 32;                 // 32 col-tiles
  const int tile = blockIdx.x * 4 + wave;   // 2048 wave-tiles total
  const int tm   = tile / NT;               // 0..63  (64-row slab)
  const int tn   = tile % NT;               // 0..31  (32-col slab)

  const _Float16* arow[4];
  const _Float16* wrow[2];
#pragma unroll
  for (int i = 0; i < 4; ++i) arow[i] = A + (size_t)(tm * 64 + i * 16 + ln) * DM;
#pragma unroll
  for (int j = 0; j < 2; ++j) wrow[j] = W + (size_t)(tn * 32 + j * 16 + ln) * DM;

  v8f acc[4][2] = {};
  for (int kk = 0; kk < DM; kk += 32) {
    v16h af[4], bf[2];
#pragma unroll
    for (int i = 0; i < 4; ++i) af[i] = load_a_frag(arow[i] + kk, lane);
#pragma unroll
    for (int j = 0; j < 2; ++j) bf[j] = load_b_frag(wrow[j] + kk, lane);
#pragma unroll
    for (int i = 0; i < 4; ++i)
#pragma unroll
      for (int j = 0; j < 2; ++j)
        acc[i][j] = WMMA_F16(af[i], bf[j], acc[i][j]);
  }

#pragma unroll
  for (int j = 0; j < 2; ++j) {
    const int col = tn * 32 + j * 16 + ln;
    const float bv = bias[col];
    const int h  = col >> 6;
    const int dk = col & 63;
#pragma unroll
    for (int i = 0; i < 4; ++i) {
#pragma unroll
      for (int r = 0; r < 8; ++r) {
        int m   = tm * 64 + i * 16 + hi + r;
        float vv = acc[i][j][r] + bv;
        if (mode == 2) {
          outf[(size_t)m * DM + col] = vv;
        } else {
          int b   = m >> 11;          // m / 2048
          int tok = m & 2047;
          size_t idx;
          if (mode == 0) idx = ((size_t)(b * NH + h) * NSEQ + tok) * DK + dk;
          else           idx = ((size_t)(b * NH + h) * DK + dk) * NSEQ + tok;
          out16[idx] = (_Float16)vv;
        }
      }
    }
  }
}

// ------------------------------------------------- flash attention per (b,h)
// Each wave owns 32 queries (two 16-row Q tiles); K/V fragments are loaded once
// per 32-key step and reused by both Q tiles (16 v_wmma per step).
// Qh, Kh: [B][H][N][DK] f16 ; Vt: [B][H][DK][N] f16 ; ctx: [B*N][DM] f16
__global__ __launch_bounds__(128) void attn_flash(const _Float16* __restrict__ Qh,
                                                  const _Float16* __restrict__ Kh,
                                                  const _Float16* __restrict__ Vt,
                                                  _Float16* __restrict__ ctx) {
  __shared__ _Float16 plds[4][2][16 * 32];  // per-wave, per-q-tile P (16q x 32k)

  const int lane = threadIdx.x & 31;
  const int wave = threadIdx.x >> 5;
  const int ln   = lane & 15;
  const int hi   = (lane >> 4) * 8;

  const int qblk = blockIdx.x & 15;         // N/128 = 16 query slabs
  const int bh   = blockIdx.x >> 4;         // b*16 + h
  const int qbase = qblk * 128 + wave * 32;

  // Q fragments: two 16-row tiles, d = 0..63 as two K=32 chunks each
  v16h q0[2], q1[2];
#pragma unroll
  for (int t = 0; t < 2; ++t) {
    const _Float16* qrow = Qh + ((size_t)bh * NSEQ + qbase + t * 16 + ln) * DK;
    q0[t] = load_a_frag(qrow, lane);
    q1[t] = load_a_frag(qrow + 32, lane);
  }

  float mrow[2][8], lrow[2][8];
  v8f o[2][4] = {};
#pragma unroll
  for (int t = 0; t < 2; ++t)
#pragma unroll
    for (int r = 0; r < 8; ++r) { mrow[t][r] = -1e30f; lrow[t][r] = 0.0f; }

  const float scale = 0.125f;               // 1/sqrt(64)

  for (int kb = 0; kb < NSEQ; kb += 32) {
    if (kb + 32 < NSEQ)
      __builtin_prefetch(Kh + ((size_t)bh * NSEQ + kb + 32 + ln) * DK, 0, 0);

    // S = Q @ K^T : two key groups of 16, shared K-frags across both Q tiles
    v8f s[2][2] = {};                       // [q tile][key group]
    {
      const _Float16* krow = Kh + ((size_t)bh * NSEQ + kb + ln) * DK;
#pragma unroll
      for (int g = 0; g < 2; ++g) {
        v16h k0 = load_b_frag(krow, lane);
        v16h k1 = load_b_frag(krow + 32, lane);
#pragma unroll
        for (int t = 0; t < 2; ++t) {
          s[t][g] = WMMA_F16(q0[t], k0, s[t][g]);
          s[t][g] = WMMA_F16(q1[t], k1, s[t][g]);
        }
        krow += 16 * DK;
      }
    }

    // online softmax per q tile (row reductions across 16 lanes of a row-half)
    float alpha[2][8];
#pragma unroll
    for (int t = 0; t < 2; ++t) {
#pragma unroll
      for (int r = 0; r < 8; ++r) {
        float a = s[t][0][r] * scale;
        float b = s[t][1][r] * scale;
        float vmax = fmaxf(a, b);
        vmax = fmaxf(vmax, __shfl_xor(vmax, 1));
        vmax = fmaxf(vmax, __shfl_xor(vmax, 2));
        vmax = fmaxf(vmax, __shfl_xor(vmax, 4));
        vmax = fmaxf(vmax, __shfl_xor(vmax, 8));
        float mnew = fmaxf(mrow[t][r], vmax);
        alpha[t][r] = __expf(mrow[t][r] - mnew);
        mrow[t][r] = mnew;
        float p0 = __expf(a - mnew);
        float p1 = __expf(b - mnew);
        s[t][0][r] = p0; s[t][1][r] = p1;
        float sum = p0 + p1;
        sum += __shfl_xor(sum, 1);
        sum += __shfl_xor(sum, 2);
        sum += __shfl_xor(sum, 4);
        sum += __shfl_xor(sum, 8);
        lrow[t][r] = lrow[t][r] * alpha[t][r] + sum;
      }
    }

    // transpose P tiles (C layout -> A layout) through per-wave LDS
#pragma unroll
    for (int t = 0; t < 2; ++t) {
      _Float16* pl = plds[wave][t];
#pragma unroll
      for (int r = 0; r < 8; ++r) {
        int rw = hi + r;
        pl[rw * 32 + ln]      = (_Float16)s[t][0][r];
        pl[rw * 32 + 16 + ln] = (_Float16)s[t][1][r];
      }
    }
    __syncthreads();
    V16H pa[2];
    {
      const int ko = (lane >> 4) * 8;
#pragma unroll
      for (int t = 0; t < 2; ++t) {
        const _Float16* pl = plds[wave][t];
        pa[t].h[0] = *(const v8h*)(pl + ln * 32 + ko);
        pa[t].h[1] = *(const v8h*)(pl + ln * 32 + 16 + ko);
      }
    }
    __syncthreads();

    // O = O*alpha + P @ V ; V-frags loaded once, reused by both q tiles
#pragma unroll
    for (int g4 = 0; g4 < 4; ++g4) {
      const _Float16* vrow = Vt + ((size_t)bh * DK + g4 * 16 + ln) * NSEQ + kb;
      v16h vb = load_b_frag(vrow, lane);
#pragma unroll
      for (int t = 0; t < 2; ++t) {
#pragma unroll
        for (int r = 0; r < 8; ++r) o[t][g4][r] *= alpha[t][r];
        o[t][g4] = WMMA_F16(pa[t].v, vb, o[t][g4]);
      }
    }
  }

  // normalize and write ctx in [b*N + q][1024] f16 layout
  const int b  = bh >> 4;
  const int hh = bh & 15;
#pragma unroll
  for (int t = 0; t < 2; ++t) {
#pragma unroll
    for (int r = 0; r < 8; ++r) {
      float inv = 1.0f / lrow[t][r];
      int qi = qbase + t * 16 + hi + r;
      size_t base = ((size_t)(b * NSEQ) + qi) * DM + hh * DK;
#pragma unroll
      for (int g = 0; g < 4; ++g)
        ctx[base + g * 16 + ln] = (_Float16)(o[t][g][r] * inv);
    }
  }
}

// ---------------------------------------------------------------------------
extern "C" void kernel_launch(void* const* d_in, const int* in_sizes, int n_in,
                              void* d_out, int out_size, void* d_ws, size_t ws_size,
                              hipStream_t stream) {
  const float* q  = (const float*)d_in[0];
  const float* k  = (const float*)d_in[1];
  const float* v  = (const float*)d_in[2];
  const float* wq = (const float*)d_in[3];
  const float* bq = (const float*)d_in[4];
  const float* wk = (const float*)d_in[5];
  const float* bk = (const float*)d_in[6];
  const float* wv = (const float*)d_in[7];
  const float* bv = (const float*)d_in[8];
  const float* wo = (const float*)d_in[9];
  const float* bo = (const float*)d_in[10];

  const size_t SZX = (size_t)MTOK * DM;   // 4,194,304 halves
  const size_t SZW = (size_t)DM * DM;     // 1,048,576 halves

  _Float16* w    = (_Float16*)d_ws;
  _Float16* xq16 = w;
  _Float16* xk16 = xq16 + SZX;
  _Float16* xv16 = xk16 + SZX;
  _Float16* wq16 = xv16 + SZX;
  _Float16* wk16 = wq16 + SZW;
  _Float16* wv16 = wk16 + SZW;
  _Float16* wo16 = wv16 + SZW;
  _Float16* Qh   = wo16 + SZW;            // [B][H][N][DK]
  _Float16* Kh   = Qh + SZX;              // [B][H][N][DK]
  _Float16* Vt   = Kh + SZX;              // [B][H][DK][N]
  _Float16* ctx  = Vt + SZX;              // [B*N][DM]

  // 1) cast everything to f16 once (whole f16 working set then lives in L2)
  cvt_f32_to_f16<<<(int)((SZX + 255) / 256), 256, 0, stream>>>(q,  xq16, (int)SZX);
  cvt_f32_to_f16<<<(int)((SZX + 255) / 256), 256, 0, stream>>>(k,  xk16, (int)SZX);
  cvt_f32_to_f16<<<(int)((SZX + 255) / 256), 256, 0, stream>>>(v,  xv16, (int)SZX);
  cvt_f32_to_f16<<<(int)((SZW + 255) / 256), 256, 0, stream>>>(wq, wq16, (int)SZW);
  cvt_f32_to_f16<<<(int)((SZW + 255) / 256), 256, 0, stream>>>(wk, wk16, (int)SZW);
  cvt_f32_to_f16<<<(int)((SZW + 255) / 256), 256, 0, stream>>>(wv, wv16, (int)SZW);
  cvt_f32_to_f16<<<(int)((SZW + 255) / 256), 256, 0, stream>>>(wo, wo16, (int)SZW);

  // 2) projections: 2048 64x32 wave-tiles, 4 waves/block -> 512 blocks
  const int gemm_blocks = (MTOK / 64) * (DM / 32) / 4;
  gemm64x32_f16<<<gemm_blocks, 128, 0, stream>>>(xq16, wq16, bq, Qh, nullptr, 0);
  gemm64x32_f16<<<gemm_blocks, 128, 0, stream>>>(xk16, wk16, bk, Kh, nullptr, 0);
  gemm64x32_f16<<<gemm_blocks, 128, 0, stream>>>(xv16, wv16, bv, Vt, nullptr, 1);

  // 3) flash attention: B*H*(N/128) = 512 blocks of 4 waves (32 queries/wave)
  attn_flash<<<BQ * NH * (NSEQ / 128), 128, 0, stream>>>(Qh, Kh, Vt, ctx);

  // 4) output projection in fp32 straight into d_out
  gemm64x32_f16<<<gemm_blocks, 128, 0, stream>>>(ctx, wo16, bo, nullptr, (float*)d_out, 2);
}
